// MambaLayer_35227321761994
// MI455X (gfx1250) — matrix operand order repaired and checked by reading, not compiled, not run on hardware
//
#include <hip/hip_runtime.h>
#include <hip/hip_bf16.h>
#include <math.h>

// ---------------- problem constants (from reference) ----------------
#define D_MODEL   1024
#define D_INNER   2048
#define D_STATE   16
#define D_CONV    4
#define DT_RANK   64
#define BATCH     4
#define SEQLEN    2048
#define NROWS     (BATCH * SEQLEN)        // 8192 token rows
#define XZ_LD     (2 * D_INNER)           // 4096
#define XDBL_LD   (DT_RANK + 2 * D_STATE) // 96

// ---------------- CDNA5 types ----------------
typedef __attribute__((ext_vector_type(16))) __bf16         v16bf;
typedef __attribute__((ext_vector_type(8)))  float          v8f;
typedef __attribute__((ext_vector_type(8)))  unsigned short ushort8;
typedef __attribute__((ext_vector_type(4)))  unsigned int   uint4v;
typedef __attribute__((ext_vector_type(8)))  int            int8v;
typedef __attribute__((ext_vector_type(4)))  int            int4v;

#if defined(__has_builtin)
#  if __has_builtin(__builtin_amdgcn_tensor_load_to_lds) && __has_builtin(__builtin_amdgcn_s_wait_tensorcnt)
#    define USE_TDM 1
#  else
#    define USE_TDM 0
#  endif
#else
#  define USE_TDM 0
#endif

__device__ __forceinline__ unsigned short f2bf(float f) {
    unsigned int u = __float_as_uint(f);
    unsigned int r = u + 0x7FFFu + ((u >> 16) & 1u);   // round-to-nearest-even
    return (unsigned short)(r >> 16);
}
__device__ __forceinline__ float bf2f(unsigned short u) {
    return __uint_as_float(((unsigned int)u) << 16);
}

// A fragment (16x32 bf16), ISA 7.12.2: elems 0..7 -> K=hi*8+j ; 8..15 -> K=16+hi*8+(j-8)
__device__ __forceinline__ v16bf frag_a(const unsigned short* rowp, int hi) {
    union { v16bf v; ushort8 u[2]; } f;
    f.u[0] = *reinterpret_cast<const ushort8*>(rowp + hi * 8);
    f.u[1] = *reinterpret_cast<const ushort8*>(rowp + 16 + hi * 8);
    return f.v;
}
// B fragment (32x16 bf16): lane -> N, elems 0..7 -> K=hi*8+j ; 8..15 -> K=16+hi*8+(j-8)
__device__ __forceinline__ v16bf frag_b(const unsigned short* rowp, int hi) {
    union { v16bf v; ushort8 u[2]; } f;
    f.u[0] = *reinterpret_cast<const ushort8*>(rowp + hi * 8);
    f.u[1] = *reinterpret_cast<const ushort8*>(rowp + 16 + hi * 8);
    return f.v;
}

#define BM 128
#define BN 128
#define BK 32
#define LDS_S 40   // padded LDS row stride in halves (80B: 16B-aligned, conflict-free)

#if USE_TDM
// Issue one TDM 2-D tile load: tile_dim = 128 x 32 bf16 elements, row stride `ld`
// elements, zero-fill rows >= rows_avail / cols >= cols_avail. LDS destination gets
// 16B of padding after every 64B row => LDS_S==40 halves row stride.
__device__ __forceinline__ void tdm_load_tile(const unsigned short* gbase, unsigned lds_off,
                                              unsigned rows_avail, unsigned cols_avail,
                                              unsigned long long stride_elems) {
    unsigned long long ga = (unsigned long long)(uintptr_t)gbase;
    uint4v g0;
    g0[0] = 1u;                                            // count=1 (valid), user mode
    g0[1] = lds_off;                                       // lds_addr (bytes)
    g0[2] = (unsigned)(ga & 0xFFFFFFFFull);                // global_addr[31:0]
    g0[3] = (unsigned)((ga >> 32) & 0x1FFFFFFull) | (2u << 30); // global_addr[56:32] | type=2
    int8v g1;
    g1[0] = (int)((1u << 16)      // data_size = 1 (2 bytes)
                | (1u << 20)      // pad_enable
                | (3u << 22)      // pad_interval: 16 DWORDs (64B) between pads
                | (3u << 25));    // pad_amount: 4 DWORDs (16B)
    g1[1] = (int)((cols_avail & 0xFFFFu) << 16);                           // tensor_dim0[15:0]
    g1[2] = (int)(((cols_avail >> 16) & 0xFFFFu) | ((rows_avail & 0xFFFFu) << 16)); // dim0 hi | dim1 lo
    g1[3] = (int)(((rows_avail >> 16) & 0xFFFFu) | (32u << 16));           // dim1 hi | tile_dim0=32
    g1[4] = (int)(128u);                                                   // tile_dim1=128, tile_dim2=0
    g1[5] = (int)(unsigned)(stride_elems & 0xFFFFFFFFull);                 // dim0_stride[31:0]
    g1[6] = (int)(unsigned)((stride_elems >> 32) & 0xFFFFull);             // dim0_stride[47:32]
    g1[7] = 0;
    int4v gz4 = {0, 0, 0, 0};
    int8v gz8 = {0, 0, 0, 0, 0, 0, 0, 0};
    __builtin_amdgcn_tensor_load_to_lds(g0, g1, gz4, gz4, gz8, 0);
}
#endif

// ---------------- bf16-WMMA GEMM: C(MxN) = A(MxK) * B(NxK)^T, A/B bf16, C f32 ----------------
__global__ __launch_bounds__(256)
void gemm_bf16_wmma(const unsigned short* __restrict__ A, const unsigned short* __restrict__ B,
                    float* __restrict__ C, unsigned short* __restrict__ Cb,
                    int M, int N, int K, int lda, int ldb, int ldc, int ldcb,
                    const float* __restrict__ bias, int act /*0=none, 1=softplus+bias*/) {
    __shared__ unsigned short sA[2][BM * LDS_S];
    __shared__ unsigned short sB[2][BM * LDS_S];

    const int t   = threadIdx.x;
    const int m0  = blockIdx.y * BM;
    const int n0  = blockIdx.x * BN;
    const int wid = t >> 5, lane = t & 31;
    const int wm  = (wid & 3) * 32;
    const int wn  = (wid >> 2) * 64;
    const int r   = lane & 15, hi = lane >> 4;

    v8f acc[2][4];
#pragma unroll
    for (int i = 0; i < 2; ++i)
#pragma unroll
        for (int j = 0; j < 4; ++j)
#pragma unroll
            for (int v = 0; v < 8; ++v) acc[i][j][v] = 0.0f;

    const int KT = K / BK;

#if USE_TDM
    if (wid == 0) {
        tdm_load_tile(A + (size_t)m0 * lda, (unsigned)(uintptr_t)&sA[0][0],
                      (unsigned)(M - m0), (unsigned)K, (unsigned long long)lda);
        tdm_load_tile(B + (size_t)n0 * ldb, (unsigned)(uintptr_t)&sB[0][0],
                      (unsigned)(N - n0), (unsigned)K, (unsigned long long)ldb);
        __builtin_amdgcn_s_wait_tensorcnt(0);
    }
#else
    ushort8 ra[2], rb[2];
    const ushort8 uz = {0, 0, 0, 0, 0, 0, 0, 0};
    auto load_tiles = [&](int k0) {
#pragma unroll
        for (int p = 0; p < 2; ++p) {
            int i = t + p * 256;                 // 512 ushort8 chunks per tile
            int row = i >> 2;
            int c8  = (i & 3) * 8;
            ra[p] = *reinterpret_cast<const ushort8*>(A + (size_t)(m0 + row) * lda + k0 + c8);
            rb[p] = (n0 + row < N)
                  ? *reinterpret_cast<const ushort8*>(B + (size_t)(n0 + row) * ldb + k0 + c8)
                  : uz;
        }
    };
    auto stash_tiles = [&](int buf) {
#pragma unroll
        for (int p = 0; p < 2; ++p) {
            int i = t + p * 256;
            int row = i >> 2;
            int c8  = (i & 3) * 8;
            *reinterpret_cast<ushort8*>(&sA[buf][row * LDS_S + c8]) = ra[p];
            *reinterpret_cast<ushort8*>(&sB[buf][row * LDS_S + c8]) = rb[p];
        }
    };
    load_tiles(0);
    stash_tiles(0);
#endif

    int cur = 0;
    for (int kt = 0; kt < KT; ++kt) {
        __syncthreads();                      // buf[cur] published to all waves
#if USE_TDM
        if (kt + 1 < KT && wid == 0) {        // DMA next tile while we compute this one
            int k0 = (kt + 1) * BK;
            tdm_load_tile(A + (size_t)m0 * lda + k0, (unsigned)(uintptr_t)&sA[cur ^ 1][0],
                          (unsigned)(M - m0), (unsigned)(K - k0), (unsigned long long)lda);
            tdm_load_tile(B + (size_t)n0 * ldb + k0, (unsigned)(uintptr_t)&sB[cur ^ 1][0],
                          (unsigned)(N - n0), (unsigned)(K - k0), (unsigned long long)ldb);
        }
#else
        if (kt + 1 < KT) load_tiles((kt + 1) * BK);
#endif

        v16bf af[2], bfrag[4];
#pragma unroll
        for (int i = 0; i < 2; ++i)
            af[i] = frag_a(&sA[cur][(wm + i * 16 + r) * LDS_S], hi);
#pragma unroll
        for (int j = 0; j < 4; ++j)
            bfrag[j] = frag_b(&sB[cur][(wn + j * 16 + r) * LDS_S], hi);

#pragma unroll
        for (int i = 0; i < 2; ++i)
#pragma unroll
            for (int j = 0; j < 4; ++j)
                acc[i][j] = __builtin_amdgcn_wmma_f32_16x16x32_bf16(
                    false, af[i], false, bfrag[j], (short)0, acc[i][j], false, false);

        if (kt + 1 < KT) {
#if USE_TDM
            if (wid == 0) __builtin_amdgcn_s_wait_tensorcnt(0);
#else
            stash_tiles(cur ^ 1);
#endif
            cur ^= 1;
        }
    }

    // C/D layout: lane -> N = lane%16 ; VGPR v -> M = v + 8*(lane>=16)
#pragma unroll
    for (int i = 0; i < 2; ++i) {
#pragma unroll
        for (int j = 0; j < 4; ++j) {
            int ncol = n0 + wn + j * 16 + r;
            if (ncol < N) {
                int mbase = m0 + wm + i * 16 + 8 * hi;
#pragma unroll
                for (int v = 0; v < 8; ++v) {
                    float val = acc[i][j][v];
                    if (act == 1) {
                        val += bias[ncol];
                        val = (val > 20.0f) ? val : log1pf(__expf(val));  // softplus
                    }
                    C[(size_t)(mbase + v) * ldc + ncol] = val;
                    if (Cb) Cb[(size_t)(mbase + v) * ldcb + ncol] = f2bf(val);
                }
            }
        }
    }
}

// ---------------- f32 -> bf16 bulk convert (4 elems/thread, packed 8B stores) ----------------
__global__ __launch_bounds__(256)
void f32_to_bf16_kernel(const float* __restrict__ in, unsigned short* __restrict__ out, int n4) {
    int i = blockIdx.x * 256 + threadIdx.x;
    if (i < n4) {
        float4 v = reinterpret_cast<const float4*>(in)[i];
        unsigned long long p = (unsigned long long)f2bf(v.x)
                             | ((unsigned long long)f2bf(v.y) << 16)
                             | ((unsigned long long)f2bf(v.z) << 32)
                             | ((unsigned long long)f2bf(v.w) << 48);
        reinterpret_cast<unsigned long long*>(out)[i] = p;
    }
}

// ---------------- causal depthwise conv(4) + bias + SiLU; writes bf16 ----------------
__global__ __launch_bounds__(256)
void conv_silu_kernel(const float* __restrict__ xz, const float* __restrict__ cw,
                      const float* __restrict__ cb, unsigned short* __restrict__ xcb) {
    int idx = blockIdx.x * 256 + threadIdx.x;
    int d = idx & (D_INNER - 1);
    int l = (idx >> 11) & (SEQLEN - 1);
    int b = idx >> 22;
    const float* base = xz + ((size_t)b * SEQLEN) * XZ_LD + d;  // x_in half of xz
    float w0 = cw[d * 4 + 0], w1 = cw[d * 4 + 1], w2 = cw[d * 4 + 2], w3 = cw[d * 4 + 3];
    float acc = cb[d];
    if (l >= 3) acc += w0 * base[(size_t)(l - 3) * XZ_LD];
    if (l >= 2) acc += w1 * base[(size_t)(l - 2) * XZ_LD];
    if (l >= 1) acc += w2 * base[(size_t)(l - 1) * XZ_LD];
    acc += w3 * base[(size_t)l * XZ_LD];
    float sig = 1.0f / (1.0f + __expf(-acc));
    xcb[(size_t)(b * SEQLEN + l) * D_INNER + d] = f2bf(acc * sig);   // SiLU
}

// ---------------- selective scan; fuses y = (scan + xc*D) * silu(z); bf16 out in-place ----------------
__global__ __launch_bounds__(256)
void scan_kernel(const unsigned short* __restrict__ xcb, const float* __restrict__ dt,
                 const float* __restrict__ xdbl, const float* __restrict__ xz,
                 const float* __restrict__ A_log, const float* __restrict__ Dp,
                 unsigned short* __restrict__ yout) {
    __shared__ float sBC[2 * D_STATE];
    const int b = blockIdx.x >> 3;
    const int d = ((blockIdx.x & 7) << 8) + threadIdx.x;

    float An[D_STATE];
#pragma unroll
    for (int n = 0; n < D_STATE; ++n) An[n] = -__expf(A_log[d * D_STATE + n]);
    const float Dd = Dp[d];

    float h[D_STATE];
#pragma unroll
    for (int n = 0; n < D_STATE; ++n) h[n] = 0.0f;

    for (int l = 0; l < SEQLEN; ++l) {
        size_t row = (size_t)(b * SEQLEN + l);
        if (threadIdx.x < 2 * D_STATE)
            sBC[threadIdx.x] = xdbl[row * XDBL_LD + DT_RANK + threadIdx.x];
        __syncthreads();

        float xv  = bf2f(xcb[row * D_INNER + d]);
        float dtv = dt[row * XZ_LD + d];              // dt overlaid on xz cols 0..2047
        float zv  = xz[row * XZ_LD + D_INNER + d];    // z half
        float dx  = dtv * xv;
        float y   = 0.0f;
#pragma unroll
        for (int n = 0; n < D_STATE; ++n) {
            float dA = __expf(dtv * An[n]);
            h[n] = h[n] * dA + dx * sBC[n];
            y += h[n] * sBC[D_STATE + n];
        }
        y += xv * Dd;
        float sig = 1.0f / (1.0f + __expf(-zv));
        yout[row * D_INNER + d] = f2bf(y * (zv * sig));
        __syncthreads();
    }
}

// ---------------- launch ----------------
extern "C" void kernel_launch(void* const* d_in, const int* in_sizes, int n_in,
                              void* d_out, int out_size, void* d_ws, size_t ws_size,
                              hipStream_t stream) {
    (void)in_sizes; (void)n_in; (void)out_size; (void)ws_size;
    const float* x          = (const float*)d_in[0];
    const float* in_proj_w  = (const float*)d_in[1];
    const float* conv_w     = (const float*)d_in[2];
    const float* conv_b     = (const float*)d_in[3];
    const float* x_proj_w   = (const float*)d_in[4];
    const float* dt_proj_w  = (const float*)d_in[5];
    const float* dt_proj_b  = (const float*)d_in[6];
    const float* A_log      = (const float*)d_in[7];
    const float* D_param    = (const float*)d_in[8];
    const float* out_proj_w = (const float*)d_in[9];
    float* out = (float*)d_out;

    char* p = (char*)d_ws;
    float* xz            = (float*)p;          p += (size_t)NROWS * XZ_LD * 4;     // 128 MB
    float* xdbl          = (float*)p;          p += (size_t)NROWS * XDBL_LD * 4;   // 3 MB
    unsigned short* xb   = (unsigned short*)p; p += (size_t)NROWS * D_MODEL * 2;   // 16 MB
    unsigned short* xcb  = (unsigned short*)p; p += (size_t)NROWS * D_INNER * 2;   // 32 MB
    unsigned short* xdblb= (unsigned short*)p; p += (size_t)NROWS * XDBL_LD * 2;   // 1.5 MB
    unsigned short* ipwb = (unsigned short*)p; p += (size_t)2 * D_INNER * D_MODEL * 2;
    unsigned short* xpwb = (unsigned short*)p; p += (size_t)XDBL_LD * D_INNER * 2;
    unsigned short* dtwb = (unsigned short*)p; p += (size_t)D_INNER * DT_RANK * 2;
    unsigned short* opwb = (unsigned short*)p; p += (size_t)D_MODEL * D_INNER * 2;
    float* dt            = xz;    // overlay dead x_in half of xz, ld = 4096
    unsigned short* yinb = xcb;   // scan writes bf16 y in-place over xcb

    // 0) one-shot f32 -> bf16 conversions (activations enter bf16 land here)
    auto cvt = [&](const float* src, unsigned short* dst, size_t n) {
        int n4 = (int)(n / 4);
        f32_to_bf16_kernel<<<(n4 + 255) / 256, 256, 0, stream>>>(src, dst, n4);
    };
    cvt(x,          xb,   (size_t)NROWS * D_MODEL);
    cvt(in_proj_w,  ipwb, (size_t)2 * D_INNER * D_MODEL);
    cvt(x_proj_w,   xpwb, (size_t)XDBL_LD * D_INNER);
    cvt(dt_proj_w,  dtwb, (size_t)D_INNER * DT_RANK);
    cvt(out_proj_w, opwb, (size_t)D_MODEL * D_INNER);

    // 1) xz = x @ in_proj_w^T            (8192 x 4096, K=1024)
    gemm_bf16_wmma<<<dim3(2 * D_INNER / BN, NROWS / BM), 256, 0, stream>>>(
        xb, ipwb, xz, nullptr, NROWS, 2 * D_INNER, D_MODEL, D_MODEL, D_MODEL, XZ_LD, 0, nullptr, 0);

    // 2) xc = silu(conv(x_in) + b)       (bf16 out)
    conv_silu_kernel<<<(NROWS * D_INNER) / 256, 256, 0, stream>>>(xz, conv_w, conv_b, xcb);

    // 3) x_dbl = xc @ x_proj_w^T         (8192 x 96, K=2048), f32 + bf16 shadow
    gemm_bf16_wmma<<<dim3((XDBL_LD + BN - 1) / BN, NROWS / BM), 256, 0, stream>>>(
        xcb, xpwb, xdbl, xdblb, NROWS, XDBL_LD, D_INNER, D_INNER, D_INNER, XDBL_LD, XDBL_LD,
        nullptr, 0);

    // 4) dt = softplus(x_dbl[:, :64] @ dt_proj_w^T + b)  (8192 x 2048, K=64) -> overlay on xz
    gemm_bf16_wmma<<<dim3(D_INNER / BN, NROWS / BM), 256, 0, stream>>>(
        xdblb, dtwb, dt, nullptr, NROWS, D_INNER, DT_RANK, XDBL_LD, DT_RANK, XZ_LD, 0,
        dt_proj_b, 1);

    // 5) selective scan + gating (bf16, in-place over xcb)
    scan_kernel<<<BATCH * (D_INNER / 256), 256, 0, stream>>>(
        xcb, dt, xdbl, xz, A_log, D_param, yinb);

    // 6) out = y @ out_proj_w^T          (8192 x 1024, K=2048)
    gemm_bf16_wmma<<<dim3(D_MODEL / BN, NROWS / BM), 256, 0, stream>>>(
        yinb, opwb, out, nullptr, NROWS, D_MODEL, D_INNER, D_INNER, D_INNER, D_MODEL, 0,
        nullptr, 0);
}
